// PaiAutoNerf_2723009266474
// MI455X (gfx1250) — matrix operand order repaired and checked by reading, compiled, not run on hardware
//
#include <hip/hip_runtime.h>
#include <stdint.h>

// ---------------------------------------------------------------------------
// PaiAutoNerf fused bf16-WMMA implementation for gfx1250 (MI455X).
// Compute-bound (~165 GFLOP/call, working set << 192MB L2). M=32 row tiles
// per workgroup with B-fragment reuse across the two 16-row sub-tiles:
// 2x WMMA per weight fetch vs the M=16 version, halving L2 weight traffic.
// ---------------------------------------------------------------------------

#define B_SZ     32
#define SIZE_N   5023
#define EYE_CNT  600
#define MAPD     40
#define LATENT_D 512
#define FEAT_TOT 640   // 512 + 64 + 64
#define MTILE    32    // rows (points) per workgroup

typedef __attribute__((ext_vector_type(16))) __bf16 v16bf;
typedef __attribute__((ext_vector_type(8)))  float  v8f;

union Frag { unsigned int u[8]; v16bf v; };

struct MLPW {
  const unsigned short* wt[10];   // bf16 transposed weights [Npad][Kpad]
  const float*          bias[10]; // original f32 biases
  int                   kp[10];   // Kpad per layer
};

// ---- helpers ---------------------------------------------------------------

__device__ __forceinline__ unsigned short f2bf(float f) {
  unsigned u = __float_as_uint(f);
  unsigned r = 0x7FFFu + ((u >> 16) & 1u);   // round-to-nearest-even
  return (unsigned short)((u + r) >> 16);
}

__device__ __forceinline__ v8f wmma_bf16(const Frag& a, const Frag& b, v8f c) {
  return __builtin_amdgcn_wmma_f32_16x16x32_bf16(
      /*neg_a=*/false, a.v, /*neg_b=*/false, b.v,
      /*c_mod=*/(short)0, c, /*reuse_a=*/false, /*reuse_b=*/false);
}

// A fragment (16x32 bf16, rows = points): lane m<16 holds row m, K {2v,2v+1}
// for v<4 and {16+2v,...} for v>=4; lanes 16-31 same rows at K+8 (per ISA
// 16-bit A layout). Source is row-major [row][k] (LDS or global), stride even.
__device__ __forceinline__ void load_a(Frag& a, const unsigned short* base,
                                       int stride, int kk, int lane) {
  const int m = lane & 15, h = lane >> 4;
  const unsigned short* row = base + (size_t)m * stride + kk + 8 * h;
#pragma unroll
  for (int v = 0; v < 4; ++v) a.u[v]     = *(const unsigned int*)(row + 2 * v);
#pragma unroll
  for (int v = 0; v < 4; ++v) a.u[4 + v] = *(const unsigned int*)(row + 16 + 2 * v);
}

// B fragment (32x16 bf16, cols = out channels): weights pre-transposed to
// [ncol][Kpad]; lane c<16 (column nt+c) reads K {0..7} and {16..23}; lanes
// 16-31 read K+8 (same K layout as A). Compiler merges into 2x global_load_b128.
__device__ __forceinline__ void load_b(Frag& b, const unsigned short* wt,
                                       int Kp, int ncol, int kk, int lane) {
  const int c = lane & 15, h = lane >> 4;
  const unsigned short* col = wt + (size_t)(ncol + c) * Kp + kk + 8 * h;
#pragma unroll
  for (int v = 0; v < 8; ++v) b.u[v] = *(const unsigned int*)(col + 2 * v + (v >= 4 ? 8 : 0));
}

// One hidden layer on a 32-row tile: out[32][W] = relu(A @ Wt + bias).
// A comes from up to two LDS regions (x-region + h-region for skip-concat).
// Waves stripe the 16-wide N tiles; each B fragment feeds both row sub-tiles.
__device__ __forceinline__ void mlp_hidden(
    const unsigned short* A0, int s0, int k0,
    const unsigned short* A1, int s1, int k1,
    const unsigned short* wt, int Kp, const float* bias, int W,
    unsigned short* out, int outStride, int wave, int lane) {
  const int c = lane & 15, h = lane >> 4;
  for (int nt = wave * 16; nt < W; nt += 64) {
    v8f acc0 = {0.f, 0.f, 0.f, 0.f, 0.f, 0.f, 0.f, 0.f};
    v8f acc1 = {0.f, 0.f, 0.f, 0.f, 0.f, 0.f, 0.f, 0.f};
    Frag a, b;
    for (int kk = 0; kk < k0; kk += 32) {
      load_b(b, wt, Kp, nt, kk, lane);
      load_a(a, A0, s0, kk, lane);                 // rows 0..15
      acc0 = wmma_bf16(a, b, acc0);
      load_a(a, A0 + 16 * s0, s0, kk, lane);       // rows 16..31
      acc1 = wmma_bf16(a, b, acc1);
    }
    for (int kk = 0; kk < k1; kk += 32) {
      load_b(b, wt, Kp, nt, k0 + kk, lane);
      load_a(a, A1, s1, kk, lane);
      acc0 = wmma_bf16(a, b, acc0);
      load_a(a, A1 + 16 * s1, s1, kk, lane);
      acc1 = wmma_bf16(a, b, acc1);
    }
    float bv = bias[nt + c];
#pragma unroll
    for (int r = 0; r < 8; ++r) {
      float v0 = fmaxf(acc0[r] + bv, 0.f);
      float v1 = fmaxf(acc1[r] + bv, 0.f);
      out[(size_t)(r + 8 * h) * outStride + nt + c]      = f2bf(v0);
      out[(size_t)(16 + r + 8 * h) * outStride + nt + c] = f2bf(v1);
    }
  }
}

// ---- weight staging --------------------------------------------------------
// src f32 [K][Nreal] row-major -> dst bf16 [Np][Kp] (transposed, K-padded).
// First K1 logical rows map to padded slots [0,K1p); remaining rows follow at
// K1p (handles the skip-concat zero gap). Pad rows/cols are zero.
__global__ void cvt_weight(const float* __restrict__ src,
                           unsigned short* __restrict__ dst,
                           int K, int Nreal, int Np, int K1, int K1p, int Kp) {
  int total = Np * Kp;
  for (int i = blockIdx.x * blockDim.x + threadIdx.x; i < total;
       i += gridDim.x * blockDim.x) {
    int nI = i / Kp, kI = i % Kp;
    int sr;
    if (kI < K1p) sr = (kI < K1) ? kI : -1;
    else { sr = K1 + (kI - K1p); if (sr >= K) sr = -1; }
    float v = (sr >= 0 && nI < Nreal) ? src[(size_t)sr * Nreal + nI] : 0.f;
    dst[i] = f2bf(v);
  }
}

__global__ void zero_f32(float* p, int n) {
  for (int i = blockIdx.x * blockDim.x + threadIdx.x; i < n;
       i += gridDim.x * blockDim.x) p[i] = 0.f;
}

__global__ void cvt_f32_bf16(const float* __restrict__ s,
                             unsigned short* __restrict__ d, int n) {
  for (int i = blockIdx.x * blockDim.x + threadIdx.x; i < n;
       i += gridDim.x * blockDim.x) d[i] = f2bf(s[i]);
}

// ---- encoder ---------------------------------------------------------------
// One workgroup = 32 points of one batch element. Input = [verts(40)|map(40)]
// padded to 96. Width 128 hidden, ping-pong LDS. Epilogue: relu + max over
// valid rows -> atomicMax(feats) (non-negative floats, so uint-max is exact).
template <int D, int SKIP, int OUTN>
__global__ __launch_bounds__(128) void enc_kernel(
    const float* __restrict__ x, const int* __restrict__ idx,
    const float* __restrict__ maps, const float* __restrict__ f1w,
    const float* __restrict__ f1b, MLPW w, float* __restrict__ feats,
    int featOff, int n) {
  constexpr int IN0P = 96, W = 128, WP = 128;
  __shared__ unsigned short sx[MTILE][IN0P];
  __shared__ unsigned short sh[2][MTILE][WP];
  const int tid = threadIdx.x, wave = tid >> 5, lane = tid & 31;
  const int tile = blockIdx.x, b = blockIdx.y;

  // Build input tile: sub = x[b, idx[p]] @ f1_w + f1_b, then map, then pad.
  for (int e = tid; e < MTILE * MAPD; e += 128) {
    int r = e / MAPD, j = e % MAPD;
    int p = tile * MTILE + r;
    float v = 0.f, mv = 0.f;
    if (p < n) {
      int gp = idx[p];
      const float* xp = x + ((size_t)b * SIZE_N + gp) * 3;
      v  = f1b[j] + xp[0] * f1w[j] + xp[1] * f1w[MAPD + j] + xp[2] * f1w[2 * MAPD + j];
      mv = maps[(size_t)p * MAPD + j];
    }
    sx[r][j]        = f2bf(v);
    sx[r][MAPD + j] = f2bf(mv);
  }
  for (int e = tid; e < MTILE * 16; e += 128) sx[e >> 4][80 + (e & 15)] = 0;
  __syncthreads();

  int cur = 0;
  for (int i = 0; i < D; ++i) {
    const unsigned short *A0, *A1 = nullptr;
    int s0, k0, s1 = 0, k1 = 0;
    if (i == 0)             { A0 = &sx[0][0];       s0 = IN0P; k0 = IN0P; }
    else if (i == SKIP + 1) { A0 = &sx[0][0];       s0 = IN0P; k0 = IN0P;
                              A1 = &sh[cur][0][0];  s1 = WP;   k1 = WP; }
    else                    { A0 = &sh[cur][0][0];  s0 = WP;   k0 = WP; }
    int nxt = (i == 0) ? 0 : (cur ^ 1);
    mlp_hidden(A0, s0, k0, A1, s1, k1, w.wt[i], w.kp[i], w.bias[i], W,
               &sh[nxt][0][0], WP, wave, lane);
    cur = nxt;
    __syncthreads();
  }

  // Output layer + relu + max-pool over the 32 rows.
  const int c = lane & 15, h = lane >> 4;
  for (int nt = wave * 16; nt < OUTN; nt += 64) {
    v8f acc0 = {0.f, 0.f, 0.f, 0.f, 0.f, 0.f, 0.f, 0.f};
    v8f acc1 = {0.f, 0.f, 0.f, 0.f, 0.f, 0.f, 0.f, 0.f};
    Frag a, b2;
    for (int kk = 0; kk < WP; kk += 32) {
      load_b(b2, w.wt[D], w.kp[D], nt, kk, lane);
      load_a(a, &sh[cur][0][0], WP, kk, lane);
      acc0 = wmma_bf16(a, b2, acc0);
      load_a(a, &sh[cur][16][0], WP, kk, lane);
      acc1 = wmma_bf16(a, b2, acc1);
    }
    float bv = w.bias[D][nt + c];
    float pm = 0.f;  // relu output >= 0, so 0 is a safe identity for max
#pragma unroll
    for (int r = 0; r < 8; ++r) {
      int p0 = tile * MTILE + r + 8 * h;
      int p1 = p0 + 16;
      float v0 = fmaxf(acc0[r] + bv, 0.f);
      float v1 = fmaxf(acc1[r] + bv, 0.f);
      if (p0 < n && v0 > pm) pm = v0;
      if (p1 < n && v1 > pm) pm = v1;
    }
    float o = __shfl_xor(pm, 16, 32);
    pm = fmaxf(pm, o);
    if (h == 0)
      atomicMax((unsigned int*)(feats + (size_t)b * FEAT_TOT + featOff + nt + c),
                __float_as_uint(pm));
  }
}

// ---- latent GEMM: z = feats(32x640) @ f2_w(640x512) + f2_b -----------------
__global__ __launch_bounds__(32) void z_kernel(
    const unsigned short* __restrict__ featsB,
    const unsigned short* __restrict__ f2wt, const float* __restrict__ f2b,
    unsigned short* __restrict__ zbf) {
  const int lane = threadIdx.x;
  const int nt = blockIdx.x * 16;
  v8f acc0 = {0.f, 0.f, 0.f, 0.f, 0.f, 0.f, 0.f, 0.f};
  v8f acc1 = {0.f, 0.f, 0.f, 0.f, 0.f, 0.f, 0.f, 0.f};
  Frag a, b;
  for (int kk = 0; kk < FEAT_TOT; kk += 32) {
    load_b(b, f2wt, FEAT_TOT, nt, kk, lane);
    load_a(a, featsB, FEAT_TOT, kk, lane);                          // rows 0-15
    acc0 = wmma_bf16(a, b, acc0);
    load_a(a, featsB + (size_t)16 * FEAT_TOT, FEAT_TOT, kk, lane);  // rows 16-31
    acc1 = wmma_bf16(a, b, acc1);
  }
  const int c = lane & 15, h = lane >> 4;
  float bv = f2b[nt + c];
#pragma unroll
  for (int r = 0; r < 8; ++r) {
    zbf[(size_t)(r + 8 * h) * LATENT_D + nt + c]      = f2bf(acc0[r] + bv);
    zbf[(size_t)(16 + r + 8 * h) * LATENT_D + nt + c] = f2bf(acc1[r] + bv);
  }
}

// ---- decoder ---------------------------------------------------------------
// Input per point = [z(512) | map(40)] padded to 576; width 144 padded to 160
// for K-stepping. Output layer produces 3 channels scattered via idx.
template <int D, int SKIP>
__global__ __launch_bounds__(128) void dec_kernel(
    const unsigned short* __restrict__ zbf, const int* __restrict__ idx,
    const float* __restrict__ maps, MLPW w, float* __restrict__ out, int n) {
  constexpr int IN0P = 576, W = 144, WP = 160;
  __shared__ unsigned short sx[MTILE][IN0P];
  __shared__ unsigned short sh[2][MTILE][WP];
  __shared__ int sIdx[MTILE];
  const int tid = threadIdx.x, wave = tid >> 5, lane = tid & 31;
  const int tile = blockIdx.x, b = blockIdx.y;

  if (tid < MTILE) {
    int p = tile * MTILE + tid;
    sIdx[tid] = (p < n) ? idx[p] : 0;
  }
  // z is identical for all 32 rows; copy as dwords.
  const unsigned int* zsrc = (const unsigned int*)(zbf + (size_t)b * LATENT_D);
  for (int e = tid; e < MTILE * 256; e += 128) {
    int r = e >> 8, k2 = e & 255;
    ((unsigned int*)&sx[r][0])[k2] = zsrc[k2];
  }
  for (int e = tid; e < MTILE * MAPD; e += 128) {
    int r = e / MAPD, j = e % MAPD;
    int p = tile * MTILE + r;
    sx[r][LATENT_D + j] = (p < n) ? f2bf(maps[(size_t)p * MAPD + j])
                                  : (unsigned short)0;
  }
  for (int e = tid; e < MTILE * 24; e += 128) sx[e / 24][552 + e % 24] = 0;
  for (int e = tid; e < 2 * MTILE * 16; e += 128) {
    int bb = e >> 9, r = (e >> 4) & 31, j = e & 15;
    sh[bb][r][W + j] = 0;   // zero the 144..159 K pad (never overwritten)
  }
  __syncthreads();

  int cur = 0;
  for (int i = 0; i < D; ++i) {
    const unsigned short *A0, *A1 = nullptr;
    int s0, k0, s1 = 0, k1 = 0;
    if (i == 0)             { A0 = &sx[0][0];       s0 = IN0P; k0 = IN0P; }
    else if (i == SKIP + 1) { A0 = &sx[0][0];       s0 = IN0P; k0 = IN0P;
                              A1 = &sh[cur][0][0];  s1 = WP;   k1 = WP; }
    else                    { A0 = &sh[cur][0][0];  s0 = WP;   k0 = WP; }
    int nxt = (i == 0) ? 0 : (cur ^ 1);
    mlp_hidden(A0, s0, k0, A1, s1, k1, w.wt[i], w.kp[i], w.bias[i], W,
               &sh[nxt][0][0], WP, wave, lane);
    cur = nxt;
    __syncthreads();
  }

  // Output layer: one 16x16 tile per row sub-tile (3 real channels).
  // Waves 0 and 1 each handle one sub-tile.
  if (wave < 2) {
    const int c = lane & 15, h = lane >> 4;
    v8f acc = {0.f, 0.f, 0.f, 0.f, 0.f, 0.f, 0.f, 0.f};
    Frag a, b2;
    for (int kk = 0; kk < WP; kk += 32) {
      load_a(a, &sh[cur][wave * 16][0], WP, kk, lane);
      load_b(b2, w.wt[D], w.kp[D], 0, kk, lane);
      acc = wmma_bf16(a, b2, acc);
    }
    float bv = (c < 3) ? w.bias[D][c] : 0.f;
#pragma unroll
    for (int r = 0; r < 8; ++r) {
      int m = wave * 16 + r + 8 * h;
      int p = tile * MTILE + m;
      if (p < n && c < 3)
        out[((size_t)b * SIZE_N + sIdx[m]) * 3 + c] = acc[r] + bv;
    }
  }
}

// ---- host ------------------------------------------------------------------
// Input leaf order assumes pytree (sorted-key) flattening of setup_inputs():
//   head_idx, head_map, leye_idx, leye_map,
//   params{ dec_head(18), dec_left_eye(8), dec_right_eye(8),
//           enc_head(18), enc_left_eye(8), enc_right_eye(8),
//           f1_b, f1_w, f2_b, f2_w },
//   reye_idx, reye_map, x
// Nerf block leaf order: b0..b_{D-1}, b_out, w0..w_{D-1}, w_out.

extern "C" void kernel_launch(void* const* d_in, const int* in_sizes, int n_in,
                              void* d_out, int out_size, void* d_ws,
                              size_t ws_size, hipStream_t stream) {
  (void)in_sizes; (void)n_in; (void)out_size; (void)ws_size;
  const int L_HEAD_IDX = 0, L_HEAD_MAP = 1, L_LEYE_IDX = 2, L_LEYE_MAP = 3;
  const int L_DEC_HEAD = 4, L_DEC_LE = 22, L_DEC_RE = 30;
  const int L_ENC_HEAD = 38, L_ENC_LE = 56, L_ENC_RE = 64;
  const int L_F1B = 72, L_F1W = 73, L_F2B = 74, L_F2W = 75;
  const int L_REYE_IDX = 76, L_REYE_MAP = 77, L_X = 78;

  char* ws = (char*)d_ws;
  float* featsF = (float*)ws;                                   // 32*640 f32
  unsigned short* featsB = (unsigned short*)(ws + B_SZ * FEAT_TOT * 4);
  unsigned short* zbf    = featsB + B_SZ * FEAT_TOT;
  unsigned short* wreg   = zbf + B_SZ * LATENT_D;
  size_t woff = 0;

  auto cvt = [&](int leaf, int K, int Nreal, int Np, int K1, int K1p, int Kp) {
    unsigned short* dst = wreg + woff;
    int total = Np * Kp;
    int blocks = (total + 255) / 256;
    if (blocks > 1024) blocks = 1024;
    cvt_weight<<<blocks, 256, 0, stream>>>((const float*)d_in[leaf], dst, K,
                                           Nreal, Np, K1, K1p, Kp);
    woff += (size_t)total;
    return (const unsigned short*)dst;
  };

  auto buildEnc = [&](int base, int D, int SKIP, int OUT) {
    MLPW m{};
    for (int i = 0; i < D; ++i) {
      int K, K1, K1p, Kp;
      if (i == 0)             { K = 80;  K1 = 80;  K1p = 96;  Kp = 96;  }
      else if (i == SKIP + 1) { K = 208; K1 = 80;  K1p = 96;  Kp = 224; }
      else                    { K = 128; K1 = 128; K1p = 128; Kp = 128; }
      m.wt[i] = cvt(base + D + 1 + i, K, 128, 128, K1, K1p, Kp);
      m.kp[i] = Kp;
      m.bias[i] = (const float*)d_in[base + i];
    }
    m.wt[D] = cvt(base + 2 * D + 1, 128, OUT, OUT, 128, 128, 128);
    m.kp[D] = 128;
    m.bias[D] = (const float*)d_in[base + D];
    return m;
  };
  auto buildDec = [&](int base, int D, int SKIP) {
    MLPW m{};
    for (int i = 0; i < D; ++i) {
      int K, K1, K1p, Kp;
      if (i == 0)             { K = 552; K1 = 552; K1p = 576; Kp = 576; }
      else if (i == SKIP + 1) { K = 696; K1 = 552; K1p = 576; Kp = 736; }
      else                    { K = 144; K1 = 144; K1p = 160; Kp = 160; }
      m.wt[i] = cvt(base + D + 1 + i, K, 144, 144, K1, K1p, Kp);
      m.kp[i] = Kp;
      m.bias[i] = (const float*)d_in[base + i];
    }
    m.wt[D] = cvt(base + 2 * D + 1, 144, 3, 16, 144, 160, 160);
    m.kp[D] = 160;
    m.bias[D] = (const float*)d_in[base + D];
    return m;
  };

  MLPW encH = buildEnc(L_ENC_HEAD, 8, 4, 512);
  MLPW encL = buildEnc(L_ENC_LE, 3, 1, 64);
  MLPW encR = buildEnc(L_ENC_RE, 3, 1, 64);
  MLPW decH = buildDec(L_DEC_HEAD, 8, 4);
  MLPW decL = buildDec(L_DEC_LE, 3, 1);
  MLPW decR = buildDec(L_DEC_RE, 3, 1);
  const unsigned short* f2wt = cvt(L_F2W, 640, 512, 512, 640, 640, 640);

  zero_f32<<<(B_SZ * FEAT_TOT + 255) / 256, 256, 0, stream>>>(featsF,
                                                              B_SZ * FEAT_TOT);

  dim3 blk(128);
  dim3 gH((SIZE_N + MTILE - 1) / MTILE, B_SZ);
  dim3 gE((EYE_CNT + MTILE - 1) / MTILE, B_SZ);

  enc_kernel<8, 4, 512><<<gH, blk, 0, stream>>>(
      (const float*)d_in[L_X], (const int*)d_in[L_HEAD_IDX],
      (const float*)d_in[L_HEAD_MAP], (const float*)d_in[L_F1W],
      (const float*)d_in[L_F1B], encH, featsF, 0, SIZE_N);
  enc_kernel<3, 1, 64><<<gE, blk, 0, stream>>>(
      (const float*)d_in[L_X], (const int*)d_in[L_LEYE_IDX],
      (const float*)d_in[L_LEYE_MAP], (const float*)d_in[L_F1W],
      (const float*)d_in[L_F1B], encL, featsF, 512, EYE_CNT);
  enc_kernel<3, 1, 64><<<gE, blk, 0, stream>>>(
      (const float*)d_in[L_X], (const int*)d_in[L_REYE_IDX],
      (const float*)d_in[L_REYE_MAP], (const float*)d_in[L_F1W],
      (const float*)d_in[L_F1B], encR, featsF, 576, EYE_CNT);

  cvt_f32_bf16<<<(B_SZ * FEAT_TOT + 255) / 256, 256, 0, stream>>>(
      featsF, featsB, B_SZ * FEAT_TOT);
  z_kernel<<<32, 32, 0, stream>>>(featsB, f2wt, (const float*)d_in[L_F2B], zbf);

  float* out = (float*)d_out;
  dec_kernel<8, 4><<<gH, blk, 0, stream>>>(
      zbf, (const int*)d_in[L_HEAD_IDX], (const float*)d_in[L_HEAD_MAP], decH,
      out, SIZE_N);
  dec_kernel<3, 1><<<gE, blk, 0, stream>>>(
      zbf, (const int*)d_in[L_LEYE_IDX], (const float*)d_in[L_LEYE_MAP], decL,
      out, EYE_CNT);
  dec_kernel<3, 1><<<gE, blk, 0, stream>>>(
      zbf, (const int*)d_in[L_REYE_IDX], (const float*)d_in[L_REYE_MAP], decR,
      out, EYE_CNT);
}
